// BiLSTM_CRF_13675175870608
// MI455X (gfx1250) — matrix operand order repaired and checked by reading, not compiled, not run on hardware
//
#include <hip/hip_runtime.h>
#include <hip/hip_bf16.h>

// ---------------------------------------------------------------------------
// BiLSTM-CRF for MI455X (gfx1250, wave32, WMMA 16x16x32 f16 + TDM async load).
// ---------------------------------------------------------------------------

typedef _Float16 h8   __attribute__((ext_vector_type(8)));
typedef _Float16 h16  __attribute__((ext_vector_type(16)));
typedef float    f8   __attribute__((ext_vector_type(8)));
typedef unsigned int u32x4 __attribute__((ext_vector_type(4)));
typedef int          i32x4 __attribute__((ext_vector_type(4)));
typedef int          i32x8 __attribute__((ext_vector_type(8)));

#define TT   512
#define BB   64
#define EE   256
#define HH   256
#define G4   1024
#define NLAB 34
#define MTOT (TT * BB)
#define NEGV (-10000.0f)
#define START_IDX 32
#define STOP_IDX  33

// LDS layout for the recurrent kernel (bytes)
#define GS_BYTES    (BB * G4 * 2)   // f16 gate buffer [col][row] : 131072
#define HS_BYTES    (BB * HH * 2)   // f16 hidden state [row][col]:  32768
#define STAGE_OFF   (GS_BYTES + HS_BYTES)   // 163840
#define STAGE_BYTES (BB * G4 * 2)   // f16 staged gin [col][row]  : 131072
#define LDS_TOTAL   (STAGE_OFF + STAGE_BYTES)  // 294912 (288 KB <= 320 KB)

// ---- WMMA operand loaders (this code's per-lane layout: contiguous k, same in A and B) --------
// A (16-bit, 16x32, MxK): lane l<16 -> row l, K = k0..k0+15 (contiguous);
//                         lane l>=16 -> row l-16, K = k0+16..k0+31.
__device__ __forceinline__ h16 load_a16(const _Float16* base, int ld, int lane, int k0) {
    const int row = lane & 15;
    const int kh  = (lane >> 4) << 4;       // 0 or 16
    const _Float16* p = base + row * ld + k0 + kh;
    h8 lo = *(const h8*)(p);                // 16B contiguous
    h8 hi = *(const h8*)(p + 8);            // 16B contiguous
    return __builtin_shufflevector(lo, hi, 0,1,2,3,4,5,6,7,8,9,10,11,12,13,14,15);
}
// B (16-bit, 32x16, KxN) with memory holding B^T row-major, i.e. W[N][K]:
// lane l -> col n = n_base + (l&15), K halves k0+16*(l>=16) .. +15 (contiguous).
__device__ __forceinline__ h16 load_b16(const _Float16* w, int ld, int lane, int n, int k0) {
    const int koff = (lane >> 4) << 4;      // 0 or 16
    return *(const h16*)(w + (size_t)n * ld + k0 + koff);
}
// Pack an f32 accumulator octet to f16 for one 16B store.
__device__ __forceinline__ h8 pack_h8(f8 a) {
    h8 r;
    #pragma unroll
    for (int i = 0; i < 8; ++i) r[i] = (_Float16)a[i];
    return r;
}

// ---- Tensor Data Mover: async 2D tile (rows x cols, f16) global -> LDS ----
// Descriptor per CDNA5 ISA 8.3/8.4. Issued by one wave; tracked by TENSORcnt.
__device__ __forceinline__ void tdm_load_2d_f16(const void* gptr, unsigned lds_off,
                                                unsigned cols, unsigned rows,
                                                unsigned row_stride_elems) {
    const unsigned long long ga = (unsigned long long)gptr;
    u32x4 g0;
    g0[0] = 1u;                                  // count=1, user-mode descriptor
    g0[1] = lds_off;                             // lds_addr (bytes)
    g0[2] = (unsigned)(ga & 0xFFFFFFFFu);        // global_addr[31:0]
    g0[3] = (unsigned)((ga >> 32) & 0x1FFFFFFu) | (2u << 30);  // addr[56:32] | type=2
    i32x8 g1;
    g1[0] = (int)(1u << 16);                     // data_size=1 (2 bytes), no flags
    g1[1] = (int)((cols & 0xFFFFu) << 16);       // tensor_dim0[15:0]
    g1[2] = (int)((cols >> 16) | ((rows & 0xFFFFu) << 16));   // dim0 hi | dim1 lo
    g1[3] = (int)((rows >> 16) | (cols << 16));  // dim1 hi | tile_dim0
    g1[4] = (int)(rows & 0xFFFFu);               // tile_dim1 | tile_dim2=0
    g1[5] = (int)row_stride_elems;               // tensor_dim0_stride[31:0]
    g1[6] = 0;                                   // stride0 hi | stride1 lo
    g1[7] = 0;                                   // stride1 hi
    i32x4 gz4 = {0, 0, 0, 0};
#if defined(__clang_major__) && (__clang_major__ >= 23)
    i32x8 gz8 = {0, 0, 0, 0, 0, 0, 0, 0};
    __builtin_amdgcn_tensor_load_to_lds(g0, g1, gz4, gz4, gz8, 0);
#else
    __builtin_amdgcn_tensor_load_to_lds(g0, g1, gz4, gz4, 0);
#endif
}

// ---------------------------------------------------------------------------
// Prep kernels
// ---------------------------------------------------------------------------
__global__ void kconvert_f16(_Float16* __restrict__ dst, const float* __restrict__ src, int n) {
    int i = blockIdx.x * blockDim.x + threadIdx.x;
    if (i < n) dst[i] = (_Float16)src[i];
}
__global__ void kzero_f16(_Float16* __restrict__ dst, int n) {
    int i = blockIdx.x * blockDim.x + threadIdx.x;
    if (i < n) dst[i] = (_Float16)0.0f;
}
__global__ void kbias(float* __restrict__ bias,
                      const float* __restrict__ bihf, const float* __restrict__ bhhf,
                      const float* __restrict__ bihb, const float* __restrict__ bhhb) {
    int i = blockIdx.x * blockDim.x + threadIdx.x;
    if (i < 2 * G4) {
        bias[i] = (i < G4) ? (bihf[i] + bhhf[i]) : (bihb[i - G4] + bhhb[i - G4]);
    }
}
// Embedding gather -> f16, layout [m = t*B + b][E]
__global__ void kembed(const int* __restrict__ sent, const float* __restrict__ embed,
                       _Float16* __restrict__ x16) {
    const int m = blockIdx.x;           // 0..MTOT-1
    const int e = threadIdx.x;          // 0..255
    const int t = m / BB, b = m % BB;
    const int tok = sent[b * TT + t];   // sentences is [B][T]
    x16[(size_t)m * EE + e] = (_Float16)embed[(size_t)tok * EE + e];
}

// ---------------------------------------------------------------------------
// K2: input-projection GEMM, output in TDM/elementwise-friendly layout:
//     gin_t[t][dir][col(1024)][b(64)] = x[t,b] . Wih_dir[col]
//     M=32768  N=1024/dir  K=256.  block=256thr(8 waves), wave covers 128 cols.
//     One m-tile (16 rows) never crosses a t boundary (64/16=4 tiles per t).
//     Packed 16B stores: the 8 accum rows are contiguous b's in gin_t.
// ---------------------------------------------------------------------------
__global__ void gemm_input(const _Float16* __restrict__ x16,   // [MTOT][256]
                           const _Float16* __restrict__ wih16, // [2][1024][256]
                           _Float16* __restrict__ gin_t)       // [T][2][1024][64]
{
    const int m0   = blockIdx.x * 16;
    const int dir  = blockIdx.y;
    const int tid  = threadIdx.x;
    const int lane = tid & 31;
    const int wave = tid >> 5;
    const int n0   = wave * 128;
    const _Float16* A = x16 + (size_t)m0 * EE;
    const _Float16* W = wih16 + (size_t)dir * G4 * EE;

    f8 acc[8];
    #pragma unroll
    for (int nt = 0; nt < 8; ++nt) acc[nt] = {};

    #pragma unroll
    for (int kk = 0; kk < 8; ++kk) {
        const int k0 = kk * 32;
        h16 a = load_a16(A, EE, lane, k0);
        h16 breg[8];
        #pragma unroll
        for (int nt = 0; nt < 8; ++nt)
            breg[nt] = load_b16(W, EE, lane, n0 + nt * 16 + (lane & 15), k0);
        #pragma unroll
        for (int nt = 0; nt < 8; ++nt)
            acc[nt] = __builtin_amdgcn_wmma_f32_16x16x32_f16(
                false, a, false, breg[nt], (short)0, acc[nt], false, false);
    }

    const int t     = m0 >> 6;                      // timestep of this tile
    const int brow0 = (m0 & 63) + ((lane >> 4) << 3);  // first of 8 consecutive b
    _Float16* dstbase = gin_t + (((size_t)t * 2 + dir) * G4) * BB;
    #pragma unroll
    for (int nt = 0; nt < 8; ++nt) {
        const int col = n0 + nt * 16 + (lane & 15);
        *(h8*)(dstbase + (size_t)col * BB + brow0) = pack_h8(acc[nt]);  // 16B store
    }
}

// ---------------------------------------------------------------------------
// K3: persistent recurrent kernel. One workgroup per direction, 1024 threads
//     (32 waves). LDS: Gs [col][row] f16 gates + Hs [row][col] f16 hidden +
//     128KB TDM stage (contiguous [col][b] slab of gin_t). c in registers.
//     TDM load of step-t gin slab overlaps the h@Whh^T WMMA GEMM.
// ---------------------------------------------------------------------------
__global__ void lstm_recurrent(const _Float16* __restrict__ gin_t,  // [T][2][1024][64]
                               const _Float16* __restrict__ whh16,  // [2][1024][256]
                               const float* __restrict__ bias,      // [2][1024]
                               const float* __restrict__ h0,        // [2][64][256]
                               const float* __restrict__ c0,        // [2][64][256]
                               _Float16* __restrict__ hs16)         // [MTOT][512]
{
    extern __shared__ char smem[];
    _Float16* Gs     = (_Float16*)smem;                    // [1024 col][64 row] f16
    _Float16* Hs     = (_Float16*)(smem + GS_BYTES);       // [64 row][256 col] f16
    _Float16* Gstage = (_Float16*)(smem + STAGE_OFF);      // [1024 col][64 row] f16

    const int dir  = blockIdx.x;
    const int tid  = threadIdx.x;
    const int lane = tid & 31;
    const int wave = tid >> 5;
    const int m0   = (wave & 3) * 16;      // 4 M-tiles cover B=64
    const int n0   = (wave >> 2) * 128;    // 8 N-groups x 128 cols = 1024
    const _Float16* W = whh16 + (size_t)dir * G4 * HH;

    // elementwise ownership: column j = tid&255, rows b = (tid>>8)*16 .. +15
    const int jcol  = tid & 255;
    const int bbase = (tid >> 8) * 16;

    const float bi = bias[dir * G4 + jcol];
    const float bf = bias[dir * G4 + 256 + jcol];
    const float bg = bias[dir * G4 + 512 + jcol];
    const float bo = bias[dir * G4 + 768 + jcol];

    float creg[16];
    #pragma unroll
    for (int u = 0; u < 16; ++u) {
        const int b = bbase + u;
        creg[u] = c0[(size_t)dir * BB * HH + b * HH + jcol];
        Hs[b * HH + jcol] = (_Float16)h0[(size_t)dir * BB * HH + b * HH + jcol];
    }
    __syncthreads();

    for (int step = 0; step < TT; ++step) {
        const int t = dir ? (TT - 1 - step) : step;

        // ---- kick off async TDM load of this step's input-gate slab -------
        // gin_t[t][dir] is one contiguous 1024x64 f16 block (128 KB)
        if (wave == 0) {
            tdm_load_2d_f16(gin_t + ((size_t)t * 2 + dir) * G4 * BB,
                            STAGE_OFF, /*cols=*/BB, /*rows=*/G4, /*stride=*/BB);
        }

        // ---- GEMM: gates = Hs @ Whh^T (per wave: 16x128 tile, K=256) ------
        f8 acc[8];
        #pragma unroll
        for (int nt = 0; nt < 8; ++nt) acc[nt] = {};
        #pragma unroll
        for (int kk = 0; kk < 8; ++kk) {
            const int k0 = kk * 32;
            h16 a = load_a16(Hs + m0 * HH, HH, lane, k0);
            h16 breg[8];
            #pragma unroll
            for (int nt = 0; nt < 8; ++nt)
                breg[nt] = load_b16(W, HH, lane, n0 + nt * 16 + (lane & 15), k0);
            #pragma unroll
            for (int nt = 0; nt < 8; ++nt)
                acc[nt] = __builtin_amdgcn_wmma_f32_16x16x32_f16(
                    false, a, false, breg[nt], (short)0, acc[nt], false, false);
        }
        // packed 16B LDS stores: 8 accum rows are 8 consecutive b's in Gs[col][row]
        {
            const int brow0 = m0 + ((lane >> 4) << 3);
            #pragma unroll
            for (int nt = 0; nt < 8; ++nt) {
                const int col = n0 + nt * 16 + (lane & 15);
                *(h8*)(Gs + col * BB + brow0) = pack_h8(acc[nt]);
            }
        }

        // TDM must have landed before anyone reads Gstage after the barrier
        if (wave == 0) __builtin_amdgcn_s_wait_tensorcnt(0);
        __syncthreads();

        // ---- elementwise gate math; vector h8 LDS reads; c in registers ---
        _Float16 gz[4][16];   // recurrent gate part, [gate][b-offset]
        _Float16 gx[4][16];   // input-projection part
        #pragma unroll
        for (int q = 0; q < 4; ++q) {
            const int col = q * 256 + jcol;
            *(h8*)&gz[q][0] = *(const h8*)(Gs + col * BB + bbase);
            *(h8*)&gz[q][8] = *(const h8*)(Gs + col * BB + bbase + 8);
            *(h8*)&gx[q][0] = *(const h8*)(Gstage + col * BB + bbase);
            *(h8*)&gx[q][8] = *(const h8*)(Gstage + col * BB + bbase + 8);
        }
        float hnew[16];
        #pragma unroll
        for (int u = 0; u < 16; ++u) {
            const float zi = (float)gz[0][u] + (float)gx[0][u] + bi;
            const float zf = (float)gz[1][u] + (float)gx[1][u] + bf;
            const float zg = (float)gz[2][u] + (float)gx[2][u] + bg;
            const float zo = (float)gz[3][u] + (float)gx[3][u] + bo;
            const float si = 1.0f / (1.0f + __expf(-zi));
            const float sf = 1.0f / (1.0f + __expf(-zf));
            const float so = 1.0f / (1.0f + __expf(-zo));
            const float c  = sf * creg[u] + si * tanhf(zg);
            creg[u] = c;
            hnew[u] = so * tanhf(c);
        }
        __syncthreads();   // all Gs/Gstage/Hs reads done before overwriting Hs

        #pragma unroll
        for (int u = 0; u < 16; ++u) {
            const int b = bbase + u;
            const _Float16 hv = (_Float16)hnew[u];
            Hs[b * HH + jcol] = hv;
            hs16[(size_t)(t * BB + b) * 512 + dir * HH + jcol] = hv;
        }
        __syncthreads();   // new h visible for next step's GEMM
    }
}

// ---------------------------------------------------------------------------
// K4: output projection  feats[m][j] = hs[m] . Wout[j] + b_out[j]
//     M=32768 K=512 N=34 (padded to 48 -> 3 waves, 1 tile each)
// ---------------------------------------------------------------------------
__global__ void gemm_out(const _Float16* __restrict__ hs16,   // [MTOT][512]
                         const _Float16* __restrict__ wout16, // [48][512]
                         const float* __restrict__ b_out,     // [34]
                         float* __restrict__ feats)           // [MTOT][34]
{
    const int m0   = blockIdx.x * 16;
    const int tid  = threadIdx.x;
    const int lane = tid & 31;
    const int wave = tid >> 5;   // 0..2
    const _Float16* A = hs16 + (size_t)m0 * 512;

    f8 acc = {};
    #pragma unroll
    for (int kk = 0; kk < 16; ++kk) {
        const int k0 = kk * 32;
        h16 a = load_a16(A, 512, lane, k0);
        const int n = wave * 16 + (lane & 15);
        h16 b = load_b16(wout16, 512, lane, n, k0);
        acc = __builtin_amdgcn_wmma_f32_16x16x32_f16(
            false, a, false, b, (short)0, acc, false, false);
    }
    const int col = wave * 16 + (lane & 15);
    if (col < NLAB) {
        const float bo = b_out[col];
        #pragma unroll
        for (int r = 0; r < 8; ++r) {
            const int m = m0 + r + ((lane >> 4) << 3);
            feats[(size_t)m * NLAB + col] = acc[r] + bo;
        }
    }
}

// ---------------------------------------------------------------------------
// K5: Viterbi DP + backtrack. One block per batch element (64 thr = 2 waves).
// ---------------------------------------------------------------------------
__global__ void viterbi(const float* __restrict__ feats,   // [T][B][34]
                        const float* __restrict__ trans,   // [34][34]
                        const unsigned char* __restrict__ masks, // [B][T]
                        int* __restrict__ parents,         // [T][B][34]
                        float* __restrict__ best,          // [B]
                        int* __restrict__ path)            // [B][T]
{
    const int b = blockIdx.x;
    const int j = threadIdx.x;
    __shared__ float dp[NLAB];
    __shared__ float nv[NLAB];

    if (j < NLAB) dp[j] = (j == START_IDX) ? 0.0f : NEGV;
    __syncthreads();

    for (int t = 0; t < TT; ++t) {
        if (j < NLAB) {
            float bestv = -3.4e38f; int arg = 0;
            #pragma unroll 2
            for (int k = 0; k < NLAB; ++k) {
                const float s = dp[k] + trans[k * NLAB + j];
                if (s > bestv) { bestv = s; arg = k; }
            }
            parents[((size_t)t * BB + b) * NLAB + j] = arg;
            const float f = feats[((size_t)t * BB + b) * NLAB + j];
            nv[j] = masks[b * TT + t] ? (bestv + f) : dp[j];
        }
        __syncthreads();
        if (j < NLAB) dp[j] = nv[j];
        __syncthreads();
    }

    if (j == 0) {
        float bv = -3.4e38f; int lab = 0;
        for (int k = 0; k < NLAB; ++k) {
            const float v = dp[k] + trans[k * NLAB + STOP_IDX];
            if (v > bv) { bv = v; lab = k; }
        }
        best[b] = bv;
        int l = lab;
        for (int t = TT - 1; t >= 0; --t) {
            path[b * TT + t] = l;
            l = parents[((size_t)t * BB + b) * NLAB + l];
        }
    }
}

// ---------------------------------------------------------------------------
// Host side
// ---------------------------------------------------------------------------
extern "C" void kernel_launch(void* const* d_in, const int* in_sizes, int n_in,
                              void* d_out, int out_size, void* d_ws, size_t ws_size,
                              hipStream_t stream) {
    (void)in_sizes; (void)n_in; (void)out_size; (void)ws_size;

    const int*   sentences = (const int*)d_in[0];
    const unsigned char* masks = (const unsigned char*)d_in[1];
    const float* embed = (const float*)d_in[2];
    const float* Wih_f = (const float*)d_in[3];
    const float* Whh_f = (const float*)d_in[4];
    const float* bih_f = (const float*)d_in[5];
    const float* bhh_f = (const float*)d_in[6];
    const float* Wih_b = (const float*)d_in[7];
    const float* Whh_b = (const float*)d_in[8];
    const float* bih_b = (const float*)d_in[9];
    const float* bhh_b = (const float*)d_in[10];
    const float* h0    = (const float*)d_in[11];
    const float* c0    = (const float*)d_in[12];
    const float* W_out = (const float*)d_in[13];
    const float* b_out = (const float*)d_in[14];
    const float* trans = (const float*)d_in[15];

    // workspace partition (256B aligned)
    char* ws = (char*)d_ws;
    size_t off = 0;
    auto take = [&](size_t bytes) -> char* {
        char* p = ws + off;
        off = (off + bytes + 255) & ~(size_t)255;
        return p;
    };
    _Float16* x16    = (_Float16*)take((size_t)MTOT * EE * 2);        // 16 MB
    _Float16* wih16  = (_Float16*)take((size_t)2 * G4 * EE * 2);      // 1 MB
    _Float16* whh16  = (_Float16*)take((size_t)2 * G4 * HH * 2);      // 1 MB
    float*    bias   = (float*)   take((size_t)2 * G4 * 4);
    _Float16* wout16 = (_Float16*)take((size_t)48 * 512 * 2);
    _Float16* gin_t  = (_Float16*)take((size_t)MTOT * 2048 * 2);      // 128 MB
    _Float16* hs16   = (_Float16*)take((size_t)MTOT * 512 * 2);       // 32 MB
    float*    feats  = (float*)   take((size_t)MTOT * NLAB * 4);      // 4.3 MB
    int*      parents= (int*)     take((size_t)MTOT * NLAB * 4);      // 4.3 MB

    float* best_out = (float*)d_out;
    int*   path_out = (int*)((float*)d_out + BB);

    // --- prep ---
    kembed<<<MTOT, 256, 0, stream>>>(sentences, embed, x16);
    const int CW = G4 * EE;       // 262144 per weight matrix
    kconvert_f16<<<(CW + 255) / 256, 256, 0, stream>>>(wih16,      Wih_f, CW);
    kconvert_f16<<<(CW + 255) / 256, 256, 0, stream>>>(wih16 + CW, Wih_b, CW);
    kconvert_f16<<<(CW + 255) / 256, 256, 0, stream>>>(whh16,      Whh_f, CW);
    kconvert_f16<<<(CW + 255) / 256, 256, 0, stream>>>(whh16 + CW, Whh_b, CW);
    kbias<<<8, 256, 0, stream>>>(bias, bih_f, bhh_f, bih_b, bhh_b);
    kzero_f16<<<(48 * 512 + 255) / 256, 256, 0, stream>>>(wout16, 48 * 512);
    kconvert_f16<<<(NLAB * 512 + 255) / 256, 256, 0, stream>>>(wout16, W_out, NLAB * 512);

    // --- input projection GEMM (all timesteps, both directions) ---
    gemm_input<<<dim3(MTOT / 16, 2), 256, 0, stream>>>(x16, wih16, gin_t);

    // --- recurrence: one persistent workgroup per direction ---
    (void)hipFuncSetAttribute(reinterpret_cast<const void*>(lstm_recurrent),
                              hipFuncAttributeMaxDynamicSharedMemorySize,
                              (int)LDS_TOTAL);
    lstm_recurrent<<<2, 1024, LDS_TOTAL, stream>>>(gin_t, whh16, bias, h0, c0, hs16);

    // --- output projection ---
    gemm_out<<<MTOT / 16, 96, 0, stream>>>(hs16, wout16, b_out, feats);

    // --- Viterbi + backtrack ---
    viterbi<<<BB, 64, 0, stream>>>(feats, trans, masks, parents, best_out, path_out);
}
